// DGCLIP_91199335564039
// MI455X (gfx1250) — matrix operand order repaired and checked by reading, not compiled, not run on hardware
//
#include <hip/hip_runtime.h>
#include <hip/hip_bf16.h>
#include <cstdint>

// ---------------------------------------------------------------------------
// DGCNN forward for MI455X (gfx1250, wave32).
//  * GEMMs: v_wmma_f32_16x16x32_f16, f32 accumulate.
//  * A (weights) tile: pre-converted f16 (padded), streamed into LDS with
//    global_load_async_to_lds_b128 + s_wait_asynccnt (CDNA5 async datapath).
//  * B tile: fused graph-feature gather (x[idx]-x, x) -> cvt -> ds_store_b128.
//  * Double-buffered LDS, one barrier per K-step.
//  * kNN top-20: fused LDS-tiled scan, no [B,N,N] matrix.
// ---------------------------------------------------------------------------

typedef __attribute__((ext_vector_type(16))) _Float16 v16h;
typedef __attribute__((ext_vector_type(8)))  _Float16 v8h;
typedef __attribute__((ext_vector_type(8)))  float    v8f;

#define BATCH 8
#define NPTS  2048
#define KNN   20
#define NK    (NPTS * KNN)   // 40960
#define BN_EPS 1e-5f

// ---------------------------------------------------------------------------
// kNN: each thread owns one query point; candidates tiled through LDS,
// scanned 4 at a time (ds_load_b128). Ascending top-20 == reference order.
// ---------------------------------------------------------------------------
template <int C>
__global__ __launch_bounds__(128) void knn_topk(const float* __restrict__ src,
                                                int srcld, size_t bstride,
                                                int* __restrict__ idx_out) {
  constexpr int T = (C <= 8) ? 512 : (C <= 64 ? 256 : 128);
  const int b = blockIdx.y;
  const float* x = src + (size_t)b * bstride;
  const int n = blockIdx.x * 128 + threadIdx.x;

  float q[C];
#pragma unroll
  for (int c = 0; c < C; ++c) q[c] = x[(size_t)c * srcld + n];

  float bestd[KNN];
  int   besti[KNN];
#pragma unroll
  for (int j = 0; j < KNN; ++j) { bestd[j] = 3.0e38f; besti[j] = 0; }

  auto upd = [&](float d, int m) {
    if (d < bestd[KNN - 1]) {
#pragma unroll
      for (int j = KNN - 1; j >= 0; --j) {
        float pd = (j > 0) ? bestd[j - 1] : -1.0f;  // sentinel < any dist
        int   pi = (j > 0) ? besti[j - 1] : 0;
        if (d < bestd[j]) {
          if (d < pd) { bestd[j] = pd; besti[j] = pi; }
          else        { bestd[j] = d;  besti[j] = m;  }
        }
      }
    }
  };

  __shared__ alignas(16) float feat[C * T];

  for (int m0 = 0; m0 < NPTS; m0 += T) {
    __syncthreads();
    for (int p = threadIdx.x; p < C * T; p += 128) {
      int cc = p / T, tt = p - cc * T;
      feat[cc * T + tt] = x[(size_t)cc * srcld + m0 + tt];
    }
    __syncthreads();
    for (int mm = 0; mm < T; mm += 4) {
      float d0 = 0.f, d1 = 0.f, d2 = 0.f, d3 = 0.f;
#pragma unroll
      for (int c = 0; c < C; ++c) {
        float4 f = *reinterpret_cast<const float4*>(&feat[c * T + mm]);
        float qc = q[c];
        float t0 = qc - f.x, t1 = qc - f.y, t2 = qc - f.z, t3 = qc - f.w;
        d0 += t0 * t0; d1 += t1 * t1; d2 += t2 * t2; d3 += t3 * t3;
      }
      upd(d0, m0 + mm);
      upd(d1, m0 + mm + 1);
      upd(d2, m0 + mm + 2);
      upd(d3, m0 + mm + 3);
    }
  }
#pragma unroll
  for (int j = 0; j < KNN; ++j)
    idx_out[((size_t)b * NPTS + n) * KNN + j] = besti[j];
}

// ---------------------------------------------------------------------------
// Weight pre-conversion: f32 [M,K] -> f16 [M,Kpad] zero-padded (so the GEMM's
// A tile is a guard-free async byte copy).
// ---------------------------------------------------------------------------
__global__ __launch_bounds__(256) void cvt_w16(const float* __restrict__ W,
                                               _Float16* __restrict__ W16,
                                               int M, int K, int Kpad) {
  int i = blockIdx.x * 256 + threadIdx.x;
  if (i >= M * Kpad) return;
  int row = i / Kpad, k = i - row * Kpad;
  W16[i] = (k < K) ? (_Float16)W[(size_t)row * K + k] : (_Float16)0.f;
}

// ---------------------------------------------------------------------------
// WMMA GEMM:  H[M, Ncols] = act(bn( W16[M,Kpad] * Bmat[K, Ncols] ))
// MODE 0 (PLAIN): Bmat dense f32, row stride srcld.
// MODE 1 (EDGE):  Bmat[r, c], n=c/20, j=c%20:
//     r <  C : src[r, idx[n,j]] - src[r, n]   (neighbor - center)
//     r >= C : src[r-C, n]                    (center)
// Tile 64x128 per 256-thread block (8 wave32s), K-step 32, double-buffered.
// A tile: 256 threads x one global_load_async_to_lds_b128 (16B each).
// ---------------------------------------------------------------------------
template <int MODE>
__global__ __launch_bounds__(256) void gemm_wmma(
    const _Float16* __restrict__ W16,  // [M, ldw] f16, zero-padded
    int ldw,
    const float* __restrict__ Bsrc,    // PLAIN: [K, Ncols]; EDGE: feature rows
    const int*   __restrict__ idx,     // EDGE: [NPTS, 20]
    int M, int Kdim, int Ncols, int Chalf, int srcld,
    const float* __restrict__ bn_g, const float* __restrict__ bn_b,
    int do_lrelu,
    float* __restrict__ Hout) {
  constexpr int LDA = 40;  // halves per row (80B stride: 16B-aligned frags)
  constexpr int LDB = 40;

  __shared__ alignas(16) _Float16 Ash[2][64 * LDA];
  __shared__ alignas(16) _Float16 Bsh[2][128 * LDB];

  const int tid  = threadIdx.x;
  const int lane = tid & 31;
  const int w    = tid >> 5;
  const int wm   = w & 3;   // 4 row-blocks of 16
  const int wn   = w >> 2;  // 2 col-blocks of 64
  const int m0   = blockIdx.y * 64;
  const int c0   = blockIdx.x * 128;

  // Staging maps (fixed for the whole kernel):
  const int am = tid >> 2;          // A: row 0..63
  const int k8 = (tid & 3) * 8;     // A: k-offset 0/8/16/24 (16B chunk)
  const int cix = tid & 127;        // B: column within tile
  const int t7  = tid >> 7;         // B: k-halves 0..15 / 16..31
  const int cg  = c0 + cix;         // B: global column

  // EDGE: per-column neighbor/center indices, hoisted out of the K loop.
  int en = 0, enb = 0;
  if (MODE == 1) {
    en = cg / KNN;
    int j = cg - en * KNN;
    enb = idx[en * KNN + j];
  }

  v8f acc[4];
#pragma unroll
  for (int t = 0; t < 4; ++t)
#pragma unroll
    for (int i = 0; i < 8; ++i) acc[t][i] = 0.f;

  const int Kpad = ((Kdim + 31) >> 5) << 5;

  int ibuf = 0;
  for (int k0 = 0; k0 < Kpad; k0 += 32, ibuf ^= 1) {
    _Float16* A_ = Ash[ibuf];
    _Float16* B_ = Bsh[ibuf];

    // ---- Stage A tile [64 x 32] via async DMA to LDS (f16 byte copy) -------
    {
      const _Float16* gp = W16 + (size_t)(m0 + am) * ldw + k0 + k8;
      unsigned lad = (unsigned)(size_t)(void*)&A_[am * LDA + k8];
      asm volatile("global_load_async_to_lds_b128 %0, %1, off"
                   :: "v"(lad), "v"(gp)
                   : "memory");
    }

    // ---- Stage B tile [32 x 128] col-major: 16 k's per thread --------------
    {
      const int rb = k0 + t7 * 16;
      _Float16 bt[16];
      if (MODE == 1) {
        if (rb + 16 <= Chalf) {          // pure neighbor-minus-center span
          float a[16], s[16];
#pragma unroll
          for (int e = 0; e < 16; ++e) a[e] = Bsrc[(size_t)(rb + e) * srcld + enb];
#pragma unroll
          for (int e = 0; e < 16; ++e) s[e] = Bsrc[(size_t)(rb + e) * srcld + en];
#pragma unroll
          for (int e = 0; e < 16; ++e) bt[e] = (_Float16)(a[e] - s[e]);
        } else if (rb >= Chalf && rb + 16 <= Kdim) {  // pure center span
          float a[16];
#pragma unroll
          for (int e = 0; e < 16; ++e)
            a[e] = Bsrc[(size_t)(rb + e - Chalf) * srcld + en];
#pragma unroll
          for (int e = 0; e < 16; ++e) bt[e] = (_Float16)a[e];
        } else {                          // ragged (only C=3 layers hit this)
#pragma unroll
          for (int e = 0; e < 16; ++e) {
            int r = rb + e;
            float v = 0.f;
            if (r < Chalf)
              v = Bsrc[(size_t)r * srcld + enb] - Bsrc[(size_t)r * srcld + en];
            else if (r < Kdim)
              v = Bsrc[(size_t)(r - Chalf) * srcld + en];
            bt[e] = (_Float16)v;
          }
        }
      } else {
        if (rb + 16 <= Kdim) {
          float a[16];
#pragma unroll
          for (int e = 0; e < 16; ++e) a[e] = Bsrc[(size_t)(rb + e) * srcld + cg];
#pragma unroll
          for (int e = 0; e < 16; ++e) bt[e] = (_Float16)a[e];
          if (k0 + 32 < Kdim)
            __builtin_prefetch(&Bsrc[(size_t)(rb + 32) * srcld + cg], 0, 1);
        } else {
#pragma unroll
          for (int e = 0; e < 16; ++e) {
            int r = rb + e;
            bt[e] = (_Float16)((r < Kdim) ? Bsrc[(size_t)r * srcld + cg] : 0.f);
          }
        }
      }
      v8h b0, b1;
#pragma unroll
      for (int e = 0; e < 8; ++e) { b0[e] = bt[e]; b1[e] = bt[8 + e]; }
      *reinterpret_cast<v8h*>(&B_[cix * LDB + t7 * 16]) = b0;
      *reinterpret_cast<v8h*>(&B_[cix * LDB + t7 * 16 + 8]) = b1;
    }

    // A DMA must have landed before the barrier releases consumers.
    asm volatile("s_wait_asynccnt 0" ::: "memory");
    __syncthreads();

    // ---- A fragment (16x32 f16): lanes 0-15: K 0-7,16-23; 16-31: 8-15,24-31
    const int amr = wm * 16 + (lane & 15);
    const int ka  = (lane < 16) ? 0 : 8;
    v8h alo = *reinterpret_cast<const v8h*>(&A_[amr * LDA + ka]);
    v8h ahi = *reinterpret_cast<const v8h*>(&A_[amr * LDA + ka + 16]);
    v16h afrag = __builtin_shufflevector(alo, ahi, 0, 1, 2, 3, 4, 5, 6, 7, 8,
                                         9, 10, 11, 12, 13, 14, 15);

    // ---- All 4 B fragments first (batch ds_loads), then 4 WMMAs -----------
    v16h bfr[4];
#pragma unroll
    for (int t = 0; t < 4; ++t) {
      const int bcol = wn * 64 + t * 16 + (lane & 15);
      const int kb   = (lane < 16) ? 0 : 8;
      v8h blo = *reinterpret_cast<const v8h*>(&B_[bcol * LDB + kb]);
      v8h bhi = *reinterpret_cast<const v8h*>(&B_[bcol * LDB + kb + 16]);
      bfr[t] = __builtin_shufflevector(blo, bhi, 0, 1, 2, 3, 4, 5, 6, 7, 8, 9,
                                       10, 11, 12, 13, 14, 15);
    }
#pragma unroll
    for (int t = 0; t < 4; ++t)
      acc[t] = __builtin_amdgcn_wmma_f32_16x16x32_f16(
          false, afrag, false, bfr[t], (short)0, acc[t], false, false);
  }

  // ---- Epilogue: eval-mode BN + leaky ReLU, f32 store -----------------------
  const float rs = rsqrtf(1.0f + BN_EPS);
#pragma unroll
  for (int t = 0; t < 4; ++t) {
    const int cb = c0 + wn * 64 + t * 16 + (lane & 15);
#pragma unroll
    for (int i = 0; i < 8; ++i) {
      const int row = m0 + wm * 16 + i + ((lane >> 4) << 3);
      float v = acc[t][i];
      if (bn_g) v = v * (bn_g[row] * rs) + bn_b[row];
      if (do_lrelu) v = (v > 0.f) ? v : 0.2f * v;
      Hout[(size_t)row * Ncols + cb] = v;
    }
  }
}

// ---------------------------------------------------------------------------
// max over k within each group of 20 columns: H[M, N*20] -> out[r*outld + n]
// ---------------------------------------------------------------------------
__global__ __launch_bounds__(256) void reduce_max_k(const float* __restrict__ H,
                                                    float* __restrict__ out,
                                                    int outld) {
  const int r = blockIdx.y;
  const int n = blockIdx.x * 256 + threadIdx.x;
  const float4* p4 =
      reinterpret_cast<const float4*>(H + (size_t)r * NK + (size_t)n * KNN);
  float m = -3.0e38f;
#pragma unroll
  for (int j = 0; j < 5; ++j) {
    float4 v = p4[j];
    m = fmaxf(m, fmaxf(fmaxf(v.x, v.y), fmaxf(v.z, v.w)));
  }
  out[(size_t)r * outld + n] = m;
}

// ---------------------------------------------------------------------------
// max over columns: H[M, Ncols] -> out[M]
// ---------------------------------------------------------------------------
__global__ __launch_bounds__(256) void colmax(const float* __restrict__ H,
                                              float* __restrict__ out,
                                              int Ncols) {
  __shared__ float red[256];
  const int r = blockIdx.x;
  float m = -3.0e38f;
  for (int c = threadIdx.x; c < Ncols; c += 256)
    m = fmaxf(m, H[(size_t)r * Ncols + c]);
  red[threadIdx.x] = m;
  __syncthreads();
  for (int s = 128; s > 0; s >>= 1) {
    if (threadIdx.x < s) red[threadIdx.x] = fmaxf(red[threadIdx.x], red[threadIdx.x + s]);
    __syncthreads();
  }
  if (threadIdx.x == 0) out[r] = red[0];
}

// ---------------------------------------------------------------------------
// Small FC: out[o] = act(bn( W[o,:].in + bias )) (+ optional 3x3 identity)
// ---------------------------------------------------------------------------
__global__ __launch_bounds__(256) void fc_fused(
    const float* __restrict__ W, const float* __restrict__ in,
    const float* __restrict__ bias, const float* __restrict__ g,
    const float* __restrict__ bt, float* __restrict__ out, int O, int K,
    int do_lrelu, int add_eye) {
  const float rs = rsqrtf(1.0f + BN_EPS);
  for (int o = threadIdx.x; o < O; o += 256) {
    float s = 0.f;
    for (int k = 0; k < K; ++k) s += W[(size_t)o * K + k] * in[k];
    if (bias) s += bias[o];
    if (g) s = s * (g[o] * rs) + bt[o];
    if (do_lrelu) s = (s > 0.f) ? s : 0.2f * s;
    if (add_eye && (o == 0 || o == 4 || o == 8)) s += 1.0f;
    out[o] = s;
  }
}

// xt[i,n] = sum_j t[i*3+j] * x[j,n]
__global__ __launch_bounds__(256) void apply_transform(
    const float* __restrict__ t9, const float* __restrict__ x,
    float* __restrict__ xt) {
  const int n = blockIdx.x * 256 + threadIdx.x;
  float x0 = x[n], x1 = x[NPTS + n], x2 = x[2 * NPTS + n];
#pragma unroll
  for (int i = 0; i < 3; ++i)
    xt[i * NPTS + n] = t9[i * 3 + 0] * x0 + t9[i * 3 + 1] * x1 + t9[i * 3 + 2] * x2;
}

// ---------------------------------------------------------------------------
extern "C" void kernel_launch(void* const* d_in, const int* in_sizes, int n_in,
                              void* d_out, int out_size, void* d_ws, size_t ws_size,
                              hipStream_t stream) {
  const float* X = (const float*)d_in[0];  // [B,3,N]
  const float* t_c1_w = (const float*)d_in[1];
  const float* t_c1_g = (const float*)d_in[2];
  const float* t_c1_b = (const float*)d_in[3];
  const float* t_c2_w = (const float*)d_in[4];
  const float* t_c2_g = (const float*)d_in[5];
  const float* t_c2_b = (const float*)d_in[6];
  const float* t_c3_w = (const float*)d_in[7];
  const float* t_c3_g = (const float*)d_in[8];
  const float* t_c3_b = (const float*)d_in[9];
  const float* t_f1_w = (const float*)d_in[10];
  const float* t_f1_g = (const float*)d_in[11];
  const float* t_f1_b = (const float*)d_in[12];
  const float* t_f2_w = (const float*)d_in[13];
  const float* t_f2_bi = (const float*)d_in[14];
  const float* t_f2_g = (const float*)d_in[15];
  const float* t_f2_b = (const float*)d_in[16];
  const float* t_f3_w = (const float*)d_in[17];
  const float* t_f3_bi = (const float*)d_in[18];
  const float* e1_w = (const float*)d_in[19];
  const float* e1_g = (const float*)d_in[20];
  const float* e1_b = (const float*)d_in[21];
  const float* e2_w = (const float*)d_in[22];
  const float* e2_g = (const float*)d_in[23];
  const float* e2_b = (const float*)d_in[24];
  const float* e3_w = (const float*)d_in[25];
  const float* e3_g = (const float*)d_in[26];
  const float* e3_b = (const float*)d_in[27];
  const float* e4_w = (const float*)d_in[28];
  const float* e4_g = (const float*)d_in[29];
  const float* e4_b = (const float*)d_in[30];
  const float* c5_w = (const float*)d_in[31];
  const float* c5_g = (const float*)d_in[32];
  const float* c5_b = (const float*)d_in[33];

  // ---- workspace carving (~90 MB; all offsets 256B-aligned) ----------------
  char* ws = (char*)d_ws;
  size_t off = 0;
  auto carve = [&](size_t bytes) { char* p = ws + off; off += (bytes + 255) & ~(size_t)255; return p; };
  int*   idxb  = (int*)  carve((size_t)BATCH * NPTS * KNN * 4);
  float* xcat  = (float*)carve((size_t)BATCH * 512 * NPTS * 4);
  float* xtr   = (float*)carve((size_t)BATCH * 3 * NPTS * 4);
  float* Hbuf  = (float*)carve((size_t)256 * NK * 4);
  float* G1    = (float*)carve((size_t)64 * NK * 4);
  float* t128  = (float*)carve((size_t)128 * NPTS * 4);
  float* v1024 = (float*)carve(1024 * 4);
  float* h512  = (float*)carve(512 * 4);
  float* h256  = (float*)carve(256 * 4);
  float* t9    = (float*)carve(64);
  // f16 padded weights (ldw = Kpad):
  _Float16* w16_tc1 = (_Float16*)carve((size_t)64 * 32 * 2);
  _Float16* w16_tc2 = (_Float16*)carve((size_t)128 * 64 * 2);
  _Float16* w16_tc3 = (_Float16*)carve((size_t)1024 * 128 * 2);
  _Float16* w16_e1  = (_Float16*)carve((size_t)64 * 32 * 2);
  _Float16* w16_e2  = (_Float16*)carve((size_t)64 * 128 * 2);
  _Float16* w16_e3  = (_Float16*)carve((size_t)128 * 128 * 2);
  _Float16* w16_e4  = (_Float16*)carve((size_t)256 * 256 * 2);
  _Float16* w16_c5  = (_Float16*)carve((size_t)1024 * 512 * 2);
  (void)ws_size; (void)in_sizes; (void)n_in; (void)out_size;

  auto cvt = [&](const float* Wf, _Float16* Wh, int M, int K, int Kpad) {
    int total = M * Kpad;
    cvt_w16<<<(total + 255) / 256, 256, 0, stream>>>(Wf, Wh, M, K, Kpad);
  };
  cvt(t_c1_w, w16_tc1, 64, 6, 32);
  cvt(t_c2_w, w16_tc2, 128, 64, 64);
  cvt(t_c3_w, w16_tc3, 1024, 128, 128);
  cvt(e1_w, w16_e1, 64, 6, 32);
  cvt(e2_w, w16_e2, 64, 128, 128);
  cvt(e3_w, w16_e3, 128, 128, 128);
  cvt(e4_w, w16_e4, 256, 256, 256);
  cvt(c5_w, w16_c5, 1024, 512, 512);

  const dim3 knn_grid(NPTS / 128, BATCH);
  const dim3 red_grid64(NPTS / 256, 64), red_grid128(NPTS / 256, 128), red_grid256(NPTS / 256, 256);

  // ================= T-net =================
  knn_topk<3><<<knn_grid, 128, 0, stream>>>(X, NPTS, (size_t)3 * NPTS, idxb);
  for (int b = 0; b < BATCH; ++b) {
    const float* xb = X + (size_t)b * 3 * NPTS;
    const int* ib = idxb + (size_t)b * NPTS * KNN;
    gemm_wmma<1><<<dim3(NK / 128, 1), 256, 0, stream>>>(
        w16_tc1, 32, xb, ib, 64, 6, NK, 3, NPTS, t_c1_g, t_c1_b, 1, G1);
    gemm_wmma<0><<<dim3(NK / 128, 2), 256, 0, stream>>>(
        w16_tc2, 64, G1, nullptr, 128, 64, NK, 0, NK, t_c2_g, t_c2_b, 1, Hbuf);
    reduce_max_k<<<red_grid128, 256, 0, stream>>>(Hbuf, t128, NPTS);
    gemm_wmma<0><<<dim3(NPTS / 128, 16), 256, 0, stream>>>(
        w16_tc3, 128, t128, nullptr, 1024, 128, NPTS, 0, NPTS, t_c3_g, t_c3_b, 1, Hbuf);
    colmax<<<1024, 256, 0, stream>>>(Hbuf, v1024, NPTS);
    fc_fused<<<1, 256, 0, stream>>>(t_f1_w, v1024, nullptr, t_f1_g, t_f1_b, h512, 512, 1024, 1, 0);
    fc_fused<<<1, 256, 0, stream>>>(t_f2_w, h512, t_f2_bi, t_f2_g, t_f2_b, h256, 256, 512, 1, 0);
    fc_fused<<<1, 256, 0, stream>>>(t_f3_w, h256, t_f3_bi, nullptr, nullptr, t9, 9, 256, 0, 1);
    apply_transform<<<NPTS / 256, 256, 0, stream>>>(t9, xb, xtr + (size_t)b * 3 * NPTS);
  }

  // ================= EdgeConv stack (outputs concatenated in xcat rows) =====
  // e1: C=3 -> 64 (xcat rows 0..63)
  knn_topk<3><<<knn_grid, 128, 0, stream>>>(xtr, NPTS, (size_t)3 * NPTS, idxb);
  for (int b = 0; b < BATCH; ++b) {
    gemm_wmma<1><<<dim3(NK / 128, 1), 256, 0, stream>>>(
        w16_e1, 32, xtr + (size_t)b * 3 * NPTS, idxb + (size_t)b * NPTS * KNN,
        64, 6, NK, 3, NPTS, e1_g, e1_b, 1, Hbuf);
    reduce_max_k<<<red_grid64, 256, 0, stream>>>(Hbuf, xcat + (size_t)b * 512 * NPTS, NPTS);
  }
  // e2: C=64 -> 64 (rows 64..127)
  knn_topk<64><<<knn_grid, 128, 0, stream>>>(xcat, NPTS, (size_t)512 * NPTS, idxb);
  for (int b = 0; b < BATCH; ++b) {
    float* srcb = xcat + (size_t)b * 512 * NPTS;
    gemm_wmma<1><<<dim3(NK / 128, 1), 256, 0, stream>>>(
        w16_e2, 128, srcb, idxb + (size_t)b * NPTS * KNN, 64, 128, NK, 64, NPTS,
        e2_g, e2_b, 1, Hbuf);
    reduce_max_k<<<red_grid64, 256, 0, stream>>>(Hbuf, srcb + (size_t)64 * NPTS, NPTS);
  }
  // e3: C=64 -> 128 (rows 128..255)
  knn_topk<64><<<knn_grid, 128, 0, stream>>>(xcat + (size_t)64 * NPTS, NPTS, (size_t)512 * NPTS, idxb);
  for (int b = 0; b < BATCH; ++b) {
    float* srcb = xcat + (size_t)b * 512 * NPTS + (size_t)64 * NPTS;
    gemm_wmma<1><<<dim3(NK / 128, 2), 256, 0, stream>>>(
        w16_e3, 128, srcb, idxb + (size_t)b * NPTS * KNN, 128, 128, NK, 64, NPTS,
        e3_g, e3_b, 1, Hbuf);
    reduce_max_k<<<red_grid128, 256, 0, stream>>>(Hbuf, srcb + (size_t)64 * NPTS, NPTS);
  }
  // e4: C=128 -> 256 (rows 256..511)
  knn_topk<128><<<knn_grid, 128, 0, stream>>>(xcat + (size_t)128 * NPTS, NPTS, (size_t)512 * NPTS, idxb);
  for (int b = 0; b < BATCH; ++b) {
    float* srcb = xcat + (size_t)b * 512 * NPTS + (size_t)128 * NPTS;
    gemm_wmma<1><<<dim3(NK / 128, 4), 256, 0, stream>>>(
        w16_e4, 256, srcb, idxb + (size_t)b * NPTS * KNN, 256, 256, NK, 128, NPTS,
        e4_g, e4_b, 1, Hbuf);
    reduce_max_k<<<red_grid256, 256, 0, stream>>>(Hbuf, srcb + (size_t)128 * NPTS, NPTS);
  }
  // conv5 (512 -> 1024) + global max pool -> d_out
  for (int b = 0; b < BATCH; ++b) {
    gemm_wmma<0><<<dim3(NPTS / 128, 16), 256, 0, stream>>>(
        w16_c5, 512, xcat + (size_t)b * 512 * NPTS, nullptr, 1024, 512, NPTS, 0,
        NPTS, c5_g, c5_b, 1, Hbuf);
    colmax<<<1024, 256, 0, stream>>>(Hbuf, (float*)d_out + (size_t)b * 1024, NPTS);
  }
}
